// GQA_764504179400
// MI455X (gfx1250) — matrix operand order repaired and checked by reading, not compiled, not run on hardware
//
#include <hip/hip_runtime.h>
#include <hip/hip_bf16.h>

// ---------------------------------------------------------------------------
// GQA forward for MI455X (gfx1250, wave32, WMMA + async global->LDS).
// Pipeline:
//   1) cvt x -> bf16; transpose-cvt Wqkv, Wproj -> K-major bf16
//   2) WMMA GEMM (async-LDS double-buffered A tile, register-pipelined B):
//      qkv_bf16 = Xb @ WqkvT (+ f32 kv_cache slice of d_out in epilogue)
//   3) build transposed K (bf16 [Hk][64][S]) for score B-fragments
//   4) flash-style block-banded attention (bf16 WMMA QK^T and PV, online
//      softmax with ALiBi + causal sliding window) -> att bf16
//   5) WMMA GEMM: out[8192,1024] = att @ WprojT  (f32 store to d_out)
// ---------------------------------------------------------------------------

typedef __bf16 bf16_t;
typedef __attribute__((ext_vector_type(16))) __bf16 v16bf;
typedef __attribute__((ext_vector_type(8)))  __bf16 v8bf;
typedef __attribute__((ext_vector_type(8)))  float  v8f;

#define S_LEN   8192
#define D_MODEL 1024
#define N_QKV   1536
#define HQ      16
#define HK      4
#define HD      64
#define WIN     512

static __device__ __forceinline__ v16bf bf_combine(v8bf lo, v8bf hi) {
  v16bf r;
#pragma unroll
  for (int i = 0; i < 8; ++i) { r[i] = lo[i]; r[i + 8] = hi[i]; }
  return r;
}

// A operand (16x32 bf16): lane L<16 holds M=L, K in {kb..kb+7, kb+16..kb+23},
// kb = 8*(L>=16). `row` points at A[M_row][k0] (global or LDS).
static __device__ __forceinline__ v16bf load_afrag(const bf16_t* row, int kb) {
  return bf_combine(*(const v8bf*)(row + kb), *(const v8bf*)(row + kb + 16));
}

// B operand (32x16 bf16, row-major K x N source): lane L holds N=L%16, K in
// {kb..kb+7, kb+16..kb+23}, kb = 8*(L>=16). `p` points at B[k0][n0+L%16].
static __device__ __forceinline__ v16bf load_bfrag(const bf16_t* p, int ld) {
  const int kb = (threadIdx.x & 16) >> 1;
  v16bf r;
#pragma unroll
  for (int i = 0; i < 8; ++i) {
    r[i] = p[(size_t)(kb + i) * ld];
    r[i + 8] = p[(size_t)(kb + 16 + i) * ld];
  }
  return r;
}

static __device__ __forceinline__ v8f wmma_bf16(v16bf a, v16bf b, v8f c) {
  return __builtin_amdgcn_wmma_f32_16x16x32_bf16(
      /*neg_a=*/false, a, /*neg_b=*/false, b,
      /*c_mod=*/(short)0, c, /*reuse_a=*/false, /*reuse_b=*/false);
}

// CDNA5 async global->LDS (tracked by ASYNCcnt).  GVS mode:
// LDS address in vdst, global address saddr + vaddr.  16 bytes per lane.
static __device__ __forceinline__ void async_load_b128(unsigned lds_off,
                                                       unsigned byte_off,
                                                       const bf16_t* sbase) {
  asm volatile("global_load_async_to_lds_b128 %0, %1, %2"
               :: "v"(lds_off), "v"(byte_off), "s"(sbase) : "memory");
}
static __device__ __forceinline__ void wait_async0() {
  asm volatile("s_wait_asynccnt 0x0" ::: "memory");
}
static __device__ __forceinline__ void wait_ds0() {
  asm volatile("s_wait_dscnt 0x0" ::: "memory");
}

// ---------------------------------------------------------------------------
// Elementwise / transpose conversion passes (bandwidth-bound)
// ---------------------------------------------------------------------------
__global__ void k_cvt_bf16(const float* __restrict__ src, bf16_t* __restrict__ dst,
                           long long n) {
  long long i = (long long)blockIdx.x * blockDim.x + threadIdx.x;
  long long stride = (long long)gridDim.x * blockDim.x;
  for (; i < n; i += stride) dst[i] = (bf16_t)src[i];
}

// src [rows][cols] f32 (row-major) -> dst [cols][rows] bf16
__global__ void k_transpose_cvt(const float* __restrict__ src, bf16_t* __restrict__ dst,
                                int rows, int cols) {
  long long n = (long long)rows * cols;
  long long i = (long long)blockIdx.x * blockDim.x + threadIdx.x;
  long long stride = (long long)gridDim.x * blockDim.x;
  for (; i < n; i += stride) {
    int r = (int)(i % rows);          // output minor = src row
    int c = (int)(i / rows);          // output major = src col
    dst[i] = (bf16_t)src[(size_t)r * cols + c];
  }
}

// Ktb[hk][d][s] = qkvb[s][ (HQ+hk)*HD + d ]
__global__ void k_build_kt(const bf16_t* __restrict__ qkvb, bf16_t* __restrict__ ktb) {
  long long n = (long long)HK * HD * S_LEN;
  long long i = (long long)blockIdx.x * blockDim.x + threadIdx.x;
  long long stride = (long long)gridDim.x * blockDim.x;
  for (; i < n; i += stride) {
    int s  = (int)(i & (S_LEN - 1));
    int d  = (int)((i >> 13) & (HD - 1));
    int hk = (int)(i >> 19);
    ktb[i] = qkvb[(size_t)s * N_QKV + (HQ + hk) * HD + d];
  }
}

// ---------------------------------------------------------------------------
// Shared WMMA GEMM mainloop.  Block = 256 thr = 8 waves (2 M x 4 N).
// Wave tile 32x64, block tile 64x256, K = 1024, lda = 1024.
// A tile (64x32 bf16, 4KB) staged in LDS via async b128, double-buffered and
// shared by all 4 N-wave columns.  B fragments register-pipelined one k-step
// ahead so their load latency hides under the 8 WMMAs of the previous step.
// ---------------------------------------------------------------------------
static __device__ __forceinline__ void gemm_mainloop(const bf16_t* __restrict__ A,
                                                     const bf16_t* __restrict__ B,
                                                     int ldb, int m0blk, int n0,
                                                     bf16_t* As /* [2][64][32] */,
                                                     v8f acc[2][4]) {
  const int tid = threadIdx.x;
  const int lane = tid & 31;
  const int wm = (tid >> 5) & 1;
  const int half8 = (lane >> 4) << 3;
  const int mrow = lane & 15;
  constexpr int NIT = D_MODEL / 32;

  // per-thread async copy slot: row = tid/4, 8-elem chunk = tid%4
  const int cr = tid >> 2, cc = (tid & 3) * 8;
  const unsigned lds_base = (unsigned)(size_t)(void*)As;
  const unsigned lds_slot = lds_base + (unsigned)((cr * 32 + cc) * 2);
  const unsigned g_off0 = (unsigned)(((m0blk + cr) * D_MODEL + cc) * 2);

  const bf16_t* a_row0 = As + (size_t)(wm * 32 + mrow) * 32;       // buf 0
  const bf16_t* a_row1 = a_row0 + 64 * 32;                          // buf 1

  async_load_b128(lds_slot, g_off0, A);                             // tile 0

  // B fragments for it = 0 (register pipeline primer)
  const bf16_t* bp0 = B + n0 + (lane & 15);
  v16bf bc0 = load_bfrag(bp0, ldb);
  v16bf bc1 = load_bfrag(bp0 + 16, ldb);
  v16bf bc2 = load_bfrag(bp0 + 32, ldb);
  v16bf bc3 = load_bfrag(bp0 + 48, ldb);

#pragma unroll 2
  for (int it = 0; it < NIT; ++it) {
    wait_async0();            // this wave's pending A-tile fill done
    __syncthreads();          // all waves' fills visible

    // next-step prefetches (clamped duplicate on the final iteration: the
    // target LDS buffer / registers are dead, so no hazard, no branch)
    const int itn = (it + 1 < NIT) ? (it + 1) : it;
    async_load_b128(lds_slot + (unsigned)(((it + 1) & 1) * 4096),
                    g_off0 + (unsigned)(itn * 64), A);
    const bf16_t* bpn = B + (size_t)(itn * 32) * ldb + n0 + (lane & 15);
    v16bf bn0 = load_bfrag(bpn, ldb);
    v16bf bn1 = load_bfrag(bpn + 16, ldb);
    v16bf bn2 = load_bfrag(bpn + 32, ldb);
    v16bf bn3 = load_bfrag(bpn + 48, ldb);

    const bf16_t* ar = (it & 1) ? a_row1 : a_row0;
    v16bf a0 = load_afrag(ar, half8);
    v16bf a1 = load_afrag(ar + 16 * 32, half8);

    acc[0][0] = wmma_bf16(a0, bc0, acc[0][0]);
    acc[1][0] = wmma_bf16(a1, bc0, acc[1][0]);
    acc[0][1] = wmma_bf16(a0, bc1, acc[0][1]);
    acc[1][1] = wmma_bf16(a1, bc1, acc[1][1]);
    acc[0][2] = wmma_bf16(a0, bc2, acc[0][2]);
    acc[1][2] = wmma_bf16(a1, bc2, acc[1][2]);
    acc[0][3] = wmma_bf16(a0, bc3, acc[0][3]);
    acc[1][3] = wmma_bf16(a1, bc3, acc[1][3]);

    bc0 = bn0; bc1 = bn1; bc2 = bn2; bc3 = bn3;   // renamed by unroll-2
  }
}

// GEMM 1: qkv = Xb[8192,1024] @ WqkvT[1024,1536]; bf16 qkv + f32 kv_cache.
__global__ __launch_bounds__(256) void k_gemm_qkv(const bf16_t* __restrict__ A,
                                                  const bf16_t* __restrict__ B,
                                                  bf16_t* __restrict__ Cb,
                                                  float* __restrict__ dkv) {
  __shared__ bf16_t As[2][64][32];
  const int lane = threadIdx.x & 31, wv = threadIdx.x >> 5;
  const int m0 = blockIdx.x * 64 + (wv & 1) * 32;
  const int n0 = blockIdx.y * 256 + (wv >> 1) * 64;
  const int half8 = (lane >> 4) << 3;

  v8f acc[2][4] = {};
  gemm_mainloop(A, B, N_QKV, blockIdx.x * 64, n0, &As[0][0][0], acc);

#pragma unroll
  for (int i = 0; i < 2; ++i)
#pragma unroll
    for (int j = 0; j < 4; ++j) {
      const int nbase = n0 + j * 16;
      const bool iskv = nbase >= D_MODEL;   // uniform: 1024 is tile-aligned
      const int n = nbase + (lane & 15);
#pragma unroll
      for (int r = 0; r < 8; ++r) {
        int m = m0 + i * 16 + r + half8;
        float c = acc[i][j][r];
        Cb[(size_t)m * N_QKV + n] = (bf16_t)c;
        if (iskv) dkv[(size_t)m * (2 * HK * HD) + (n - D_MODEL)] = c;
      }
    }
}

// GEMM 2: out = Ab[8192,1024] @ WprojT[1024,1024], f32 store.
__global__ __launch_bounds__(256) void k_gemm_out(const bf16_t* __restrict__ A,
                                                  const bf16_t* __restrict__ B,
                                                  float* __restrict__ C) {
  __shared__ bf16_t As[2][64][32];
  const int lane = threadIdx.x & 31, wv = threadIdx.x >> 5;
  const int m0 = blockIdx.x * 64 + (wv & 1) * 32;
  const int n0 = blockIdx.y * 256 + (wv >> 1) * 64;
  const int half8 = (lane >> 4) << 3;

  v8f acc[2][4] = {};
  gemm_mainloop(A, B, D_MODEL, blockIdx.x * 64, n0, &As[0][0][0], acc);

#pragma unroll
  for (int i = 0; i < 2; ++i)
#pragma unroll
    for (int j = 0; j < 4; ++j)
#pragma unroll
      for (int r = 0; r < 8; ++r) {
        int m = m0 + i * 16 + r + half8;
        int n = n0 + j * 16 + (lane & 15);
        C[(size_t)m * D_MODEL + n] = acc[i][j][r];
      }
}

// ---------------------------------------------------------------------------
// Block-banded sliding-window attention, flash-style, one 16-query tile per
// wave.  Per 32-key chunk: 4 WMMA scores (K=64 head dim), online softmax with
// ALiBi+causal-window masking, P re-layout through LDS, 4 WMMA for PV (N=64).
// ---------------------------------------------------------------------------
__global__ __launch_bounds__(128) void k_attn(const bf16_t* __restrict__ qkvb,
                                              const bf16_t* __restrict__ ktb,
                                              bf16_t* __restrict__ attb) {
  __shared__ bf16_t pbuf[4][16][32];  // per-wave 16x32 P tile (bf16)

  const int lane = threadIdx.x & 31, wv = threadIdx.x >> 5;
  const int wid = blockIdx.x * 4 + wv;        // 8192 q-tiles total
  const int qt  = wid & 31;                   // 16-query tile within block
  const int h   = (wid >> 5) & 15;            // q head
  const int blk = wid >> 9;                   // attention block (of 512)
  const int hk  = h >> 2;                     // GQA kv head
  const float slope = exp2f(-0.5f * (float)(h + 1));  // ALiBi
  const int half8 = (lane >> 4) << 3;
  const int nlane = lane & 15;

  // Q fragments (16x64 -> two 16x32 A operands), rows strided by qkv pitch
  const bf16_t* qrow =
      qkvb + (size_t)(blk * WIN + qt * 16 + nlane) * N_QKV + h * HD;
  const v16bf qa0 = load_afrag(qrow, half8);
  const v16bf qa1 = load_afrag(qrow + 32, half8);

  v8f acc[4] = {};
  float rowmax[8], rowsum[8];
#pragma unroll
  for (int r = 0; r < 8; ++r) { rowmax[r] = -3.0e38f; rowsum[r] = 0.0f; }

  // valid 32-key chunks: u >= qt*16 (window) and u >= 512 for block 0
  const int uu_lo = (blk == 0) ? 16 : (qt >> 1);
  int uu_hi = (qt * 16 + 527) >> 5; if (uu_hi > 31) uu_hi = 31;

  const bf16_t* ktbase = ktb + (size_t)hk * HD * S_LEN;
  const int voff = (HQ + HK + hk) * HD;

  for (int uu = uu_lo; uu <= uu_hi; ++uu) {
    const int u0  = uu * 32;                  // context pos (0..1023)
    const int kg0 = blk * WIN - WIN + u0;     // global key index of chunk

    // ---- scores: two 16x16 tiles over 32 keys, K-depth 64 (head dim) ----
    v8f s0 = {}, s1 = {};
    {
      const bf16_t* kt0 = ktbase + kg0 + nlane;                       // d 0..31
      const bf16_t* kt1 = ktbase + (size_t)32 * S_LEN + kg0 + nlane;  // d 32..63
      s0 = wmma_bf16(qa0, load_bfrag(kt0, S_LEN), s0);
      s0 = wmma_bf16(qa1, load_bfrag(kt1, S_LEN), s0);
      s1 = wmma_bf16(qa0, load_bfrag(kt0 + 16, S_LEN), s1);
      s1 = wmma_bf16(qa1, load_bfrag(kt1 + 16, S_LEN), s1);
    }

    // ---- online softmax with ALiBi + causal window mask ----
    float fr8[8];
#pragma unroll
    for (int r = 0; r < 8; ++r) {
      const int mloc = r + half8;
      const int qpos = qt * 16 + mloc;
      const int ue0 = u0 + nlane, ue1 = ue0 + 16;
      const int d0 = qpos + WIN - ue0, d1 = qpos + WIN - ue1;
      const bool ok0 = (d0 >= 0) & (d0 <= WIN);
      const bool ok1 = (d1 >= 0) & (d1 <= WIN);
      float b0 = ok0 ? (s0[r] * 0.125f - slope * (float)d0) : -3.0e38f;
      float b1 = ok1 ? (s1[r] * 0.125f - slope * (float)d1) : -3.0e38f;
      float tm = fmaxf(b0, b1);
      tm = fmaxf(tm, __shfl_xor(tm, 1));
      tm = fmaxf(tm, __shfl_xor(tm, 2));
      tm = fmaxf(tm, __shfl_xor(tm, 4));
      tm = fmaxf(tm, __shfl_xor(tm, 8));
      const float nm = fmaxf(rowmax[r], tm);
      const float fr = __expf(rowmax[r] - nm);
      const float p0 = ok0 ? __expf(b0 - nm) : 0.0f;
      const float p1 = ok1 ? __expf(b1 - nm) : 0.0f;
      float ps = p0 + p1;
      ps += __shfl_xor(ps, 1);
      ps += __shfl_xor(ps, 2);
      ps += __shfl_xor(ps, 4);
      ps += __shfl_xor(ps, 8);
      rowsum[r] = rowsum[r] * fr + ps;
      rowmax[r] = nm;
      fr8[r] = fr;
      pbuf[wv][mloc][nlane]      = (bf16_t)p0;   // C-layout -> LDS
      pbuf[wv][mloc][16 + nlane] = (bf16_t)p1;
    }
#pragma unroll
    for (int dt = 0; dt < 4; ++dt)
#pragma unroll
      for (int r = 0; r < 8; ++r) acc[dt][r] *= fr8[r];

    // wave-private LDS round-trip: C-layout P -> A-layout P
    wait_ds0();
    const v16bf pa = load_afrag(&pbuf[wv][nlane][0], half8);

    // ---- PV: 16x32 P x 32x64 V ----
#pragma unroll
    for (int dt = 0; dt < 4; ++dt) {
      const bf16_t* vp = qkvb + (size_t)kg0 * N_QKV + voff + dt * 16 + nlane;
      acc[dt] = wmma_bf16(pa, load_bfrag(vp, N_QKV), acc[dt]);
    }
  }

  // ---- normalize + store bf16 att [S, HQ*HD] ----
  float inv[8];
#pragma unroll
  for (int r = 0; r < 8; ++r) inv[r] = 1.0f / rowsum[r];
  const size_t srow = (size_t)(blk * WIN + qt * 16);
#pragma unroll
  for (int dt = 0; dt < 4; ++dt)
#pragma unroll
    for (int r = 0; r < 8; ++r)
      attb[(srow + r + half8) * D_MODEL + h * HD + dt * 16 + nlane] =
          (bf16_t)(acc[dt][r] * inv[r]);
}

// ---------------------------------------------------------------------------
extern "C" void kernel_launch(void* const* d_in, const int* in_sizes, int n_in,
                              void* d_out, int out_size, void* d_ws, size_t ws_size,
                              hipStream_t stream) {
  const float* x     = (const float*)d_in[0];  // [8192, 1024]
  const float* Wqkv  = (const float*)d_in[1];  // [1536, 1024]
  const float* Wproj = (const float*)d_in[2];  // [1024, 1024]
  float* out = (float*)d_out;                          // [8192, 1024]
  float* dkv = out + (size_t)S_LEN * D_MODEL;          // [8192, 8, 64]

  char* ws = (char*)d_ws;
  bf16_t* Xb     = (bf16_t*)ws; ws += (size_t)S_LEN * D_MODEL * 2;   // 16 MB
  bf16_t* WqkvT  = (bf16_t*)ws; ws += (size_t)D_MODEL * N_QKV * 2;   //  3 MB
  bf16_t* WprojT = (bf16_t*)ws; ws += (size_t)D_MODEL * D_MODEL * 2; //  2 MB
  bf16_t* qkvb   = (bf16_t*)ws; ws += (size_t)S_LEN * N_QKV * 2;     // 24 MB
  bf16_t* Ktb    = (bf16_t*)ws; ws += (size_t)HK * HD * S_LEN * 2;   //  4 MB
  bf16_t* attb   = (bf16_t*)ws;                                      // 16 MB

  // 1) conversions / weight transposes
  k_cvt_bf16<<<4096, 256, 0, stream>>>(x, Xb, (long long)S_LEN * D_MODEL);
  k_transpose_cvt<<<2048, 256, 0, stream>>>(Wqkv, WqkvT, N_QKV, D_MODEL);
  k_transpose_cvt<<<2048, 256, 0, stream>>>(Wproj, WprojT, D_MODEL, D_MODEL);

  // 2) fused QKV projection (+ f32 kv_cache epilogue)
  k_gemm_qkv<<<dim3(S_LEN / 64, N_QKV / 256), 256, 0, stream>>>(Xb, WqkvT, qkvb, dkv);

  // 3) transposed-K for score B operands
  k_build_kt<<<2048, 256, 0, stream>>>(qkvb, Ktb);

  // 4) attention: 8192 q-tiles, 4 waves per 128-thread block
  k_attn<<<2048, 128, 0, stream>>>(qkvb, Ktb, attb);

  // 5) output projection
  k_gemm_out<<<dim3(S_LEN / 64, D_MODEL / 256), 256, 0, stream>>>(attb, WprojT, out);
}
